// DLRM_49744311222349
// MI455X (gfx1250) — matrix operand order repaired and checked by reading, not compiled, not run on hardware
//
#include <hip/hip_runtime.h>
#include <hip/hip_bf16.h>

// ---------------------------------------------------------------------------
// DLRM forward for MI455X (gfx1250, wave32, WMMA).
// Heavy op: p0 = (z zT flattened) @ pw0  ==  sum_i diag(z[:,i]) * (Z @ W_i)
// implemented with v_wmma_f32_16x16x32_bf16, weights staged fp32->bf16 via LDS.
// ---------------------------------------------------------------------------

typedef __attribute__((ext_vector_type(16))) __bf16 v16bf;
typedef __attribute__((ext_vector_type(8)))  float  v8f;

#define BATCH   512
#define KCAT    432      // 16 dense + 26*16 sparse
#define KPAD    448      // padded to 14 * 32
#define LDSROW  456      // KPAD + 8 u16 pad: 228 dwords/row -> conflict-free b128
#define NPRED0  512
#define CARD    100000

static __device__ __forceinline__ unsigned int f2bf(float f) {
    unsigned int u = __float_as_uint(f);
    unsigned int r = u + 0x7FFFu + ((u >> 16) & 1u);   // round-to-nearest-even
    return r >> 16;
}

static __device__ __forceinline__ v8f vzero8() {
    v8f v = {0.f,0.f,0.f,0.f,0.f,0.f,0.f,0.f};
    return v;
}

// 16 bf16 elements gathered as two 16-byte chunks (global_load_b128 / ds_load_b128)
static __device__ __forceinline__ v16bf load_frag16(const unsigned short* p0,
                                                    const unsigned short* p1) {
    union { uint4 u[2]; v16bf v; } f;
    f.u[0] = *(const uint4*)p0;
    f.u[1] = *(const uint4*)p1;
    return f.v;
}

// ---------------------------------------------------------------------------
// Generic tiny GEMM + bias (+ReLU), y[B,N] = act(x[B,K] @ w[K,N] + b[N]).
// Used only for the small MLP layers (<0.2% of total FLOPs).
// ---------------------------------------------------------------------------
__global__ void k_linrelu(const float* __restrict__ x, const float* __restrict__ w,
                          const float* __restrict__ bias, float* __restrict__ y,
                          int B, int K, int N, int relu) {
    int idx = blockIdx.x * blockDim.x + threadIdx.x;
    if (idx >= B * N) return;
    int b = idx / N, n = idx % N;
    float s = bias[n];
    const float* xr = x + (size_t)b * K;
    for (int k = 0; k < K; ++k) s = fmaf(xr[k], w[(size_t)k * N + n], s);
    y[idx] = (relu && s < 0.f) ? 0.f : s;
}

// ---------------------------------------------------------------------------
// Build z = concat(dense_out, embedding gathers) in fp32 and bf16 (K padded
// to 448 with zeros so WMMA K-tail contributes exactly 0).
// ---------------------------------------------------------------------------
__global__ void k_concat(const float* __restrict__ h2, const float* __restrict__ dw3,
                         const float* __restrict__ db3, const int* __restrict__ sparse,
                         const float* __restrict__ emb,
                         float* __restrict__ z, unsigned short* __restrict__ zb) {
    int idx = blockIdx.x * blockDim.x + threadIdx.x;
    if (idx >= BATCH * KPAD) return;
    int b = idx / KPAD, c = idx % KPAD;
    float v = 0.f;
    if (c < 16) {                      // dense_out = h2 @ dw3 + db3
        float s = db3[c];
        const float* hr = h2 + (size_t)b * 64;
        for (int k = 0; k < 64; ++k) s = fmaf(hr[k], dw3[k * 16 + c], s);
        v = s;
    } else if (c < KCAT) {             // embedding gather with modulus hash
        int t = (c - 16) >> 4, e = (c - 16) & 15;
        int id = (sparse[b * 26 + t] + 1) % CARD;
        v = emb[((size_t)t * CARD + id) * 16 + e];
    }
    if (c < KCAT) z[(size_t)b * KCAT + c] = v;
    zb[(size_t)b * KPAD + c] = (unsigned short)f2bf(v);
}

__global__ void k_zero(float* __restrict__ p, int n) {
    int idx = blockIdx.x * blockDim.x + threadIdx.x;
    if (idx < n) p[idx] = 0.f;
}

// ---------------------------------------------------------------------------
// Interaction GEMM: C[512,512] += sum over this block's i-slice of
//   diag(z[:,i]) * (Z_tile @ W_i_tile), bf16 WMMA with f32 accumulate.
// grid = (N/64, M/128, 432/16), block = 256 threads = 8 waves.
// Wave w owns rows m0 = by*128 + w*16, computes 4 N-subtiles of 16.
// ---------------------------------------------------------------------------
__global__ void __launch_bounds__(256)
k_interact(const unsigned short* __restrict__ zb, const float* __restrict__ zf,
           const float* __restrict__ pw0, float* __restrict__ C) {
    __shared__ __align__(16) unsigned short Bt[64][LDSROW];  // W_i^T tile, bf16, 58 KB

    const int n0   = blockIdx.x * 64;
    const int m0b  = blockIdx.y * 128;
    const int i0   = blockIdx.z * 16;
    const int tid  = threadIdx.x;
    const int wave = tid >> 5;
    const int lane = tid & 31;
    const int half = lane >> 4;     // K-half selector in WMMA layouts
    const int lrow = lane & 15;     // M (for A/C) or N (for B) within tile
    const int m0   = m0b + wave * 16;

    // zero LDS pad rows once (avoid NaN * 0 poisoning; A pad is already 0)
    for (int x = tid; x < 64 * 16; x += 256) Bt[x >> 4][KCAT + (x & 15)] = 0;

    v8f acc[4];
    #pragma unroll
    for (int t = 0; t < 4; ++t) acc[t] = vzero8();

    for (int ii = 0; ii < 16; ++ii) {
        const int i = i0 + ii;

        // ---- stage W_i tile (432 x 64 fp32) -> Bt[n][j] bf16, transposed ----
        // Each unit = 4x4 sub-block: 4 coalesced float4 loads + 4 packed b64 stores.
        for (int u = tid; u < 108 * 16; u += 256) {
            const int jq = u >> 4;            // j-quad 0..107  (j = 4*jq)
            const int nq = (u & 15) << 2;     // col offset 0,4,...,60
            const float* wp = pw0 + ((size_t)i * KCAT + 4 * jq) * NPRED0 + n0 + nq;
            float4 r0 = *(const float4*)(wp);
            float4 r1 = *(const float4*)(wp + NPRED0);
            float4 r2 = *(const float4*)(wp + 2 * NPRED0);
            float4 r3 = *(const float4*)(wp + 3 * NPRED0);
            uint2 p;
            p.x = f2bf(r0.x) | (f2bf(r1.x) << 16);
            p.y = f2bf(r2.x) | (f2bf(r3.x) << 16);
            *(uint2*)&Bt[nq + 0][4 * jq] = p;
            p.x = f2bf(r0.y) | (f2bf(r1.y) << 16);
            p.y = f2bf(r2.y) | (f2bf(r3.y) << 16);
            *(uint2*)&Bt[nq + 1][4 * jq] = p;
            p.x = f2bf(r0.z) | (f2bf(r1.z) << 16);
            p.y = f2bf(r2.z) | (f2bf(r3.z) << 16);
            *(uint2*)&Bt[nq + 2][4 * jq] = p;
            p.x = f2bf(r0.w) | (f2bf(r1.w) << 16);
            p.y = f2bf(r2.w) | (f2bf(r3.w) << 16);
            *(uint2*)&Bt[nq + 3][4 * jq] = p;
        }
        __syncthreads();

        // ---- tmp = Z_rows @ W_i over K = 448 (14 WMMA k-steps) ----
        v8f tmp[4];
        #pragma unroll
        for (int t = 0; t < 4; ++t) tmp[t] = vzero8();

        #pragma unroll
        for (int k = 0; k < 14; ++k) {
            // A frag (16x32 bf16): lane row m0+lrow; elems 0-7 @K=32k+8*half,
            // elems 8-15 @K=32k+16+8*half  (hoisted to persistent VGPRs by compiler)
            const unsigned short* ar =
                zb + (size_t)(m0 + lrow) * KPAD + 32 * k + 8 * half;
            v16bf a = load_frag16(ar, ar + 16);
            #pragma unroll
            for (int t = 0; t < 4; ++t) {
                // B frag (32x16 bf16): lane col = t*16+lrow, elems 0-7 @K=32k+8*half, 8-15 @K=32k+16+8*half
                const unsigned short* br = &Bt[t * 16 + lrow][32 * k + 8 * half];
                v16bf b = load_frag16(br, br + 16);
                tmp[t] = __builtin_amdgcn_wmma_f32_16x16x32_bf16(
                    false, a, false, b, (short)0, tmp[t], false, false);
            }
        }

        // ---- fold: acc += z[m, i] * tmp   (m = m0 + 8*half + r per C layout) ----
        #pragma unroll
        for (int r = 0; r < 8; ++r) {
            float s = zf[(size_t)(m0 + 8 * half + r) * KCAT + i];
            #pragma unroll
            for (int t = 0; t < 4; ++t) acc[t][r] = fmaf(s, tmp[t][r], acc[t][r]);
        }
        __syncthreads();   // Bt reused next i
    }

    // ---- commit partial C tile (i-split across grid.z -> atomic add) ----
    #pragma unroll
    for (int t = 0; t < 4; ++t) {
        #pragma unroll
        for (int r = 0; r < 8; ++r) {
            atomicAdd(&C[(size_t)(m0 + 8 * half + r) * NPRED0 + n0 + t * 16 + lrow],
                      acc[t][r]);
        }
    }
}

__global__ void k_biasrelu(float* __restrict__ p, const float* __restrict__ bias, int B, int N) {
    int idx = blockIdx.x * blockDim.x + threadIdx.x;
    if (idx >= B * N) return;
    float v = p[idx] + bias[idx % N];
    p[idx] = v < 0.f ? 0.f : v;
}

__global__ void k_final(const float* __restrict__ p1, const float* __restrict__ pw2,
                        const float* __restrict__ pb2, float* __restrict__ out) {
    int b = blockIdx.x * blockDim.x + threadIdx.x;
    if (b >= BATCH) return;
    float s = pb2[0];
    const float* pr = p1 + (size_t)b * 256;
    for (int k = 0; k < 256; ++k) s = fmaf(pr[k], pw2[k], s);
    out[b] = 1.f / (1.f + __expf(-s));
}

// ---------------------------------------------------------------------------
extern "C" void kernel_launch(void* const* d_in, const int* in_sizes, int n_in,
                              void* d_out, int out_size, void* d_ws, size_t ws_size,
                              hipStream_t stream) {
    const float* dense  = (const float*)d_in[0];
    const int*   sparse = (const int*)  d_in[1];
    const float* emb    = (const float*)d_in[2];
    const float* dw0 = (const float*)d_in[3];  const float* db0 = (const float*)d_in[4];
    const float* dw1 = (const float*)d_in[5];  const float* db1 = (const float*)d_in[6];
    const float* dw2 = (const float*)d_in[7];  const float* db2 = (const float*)d_in[8];
    const float* dw3 = (const float*)d_in[9];  const float* db3 = (const float*)d_in[10];
    const float* pw0 = (const float*)d_in[11]; const float* pb0 = (const float*)d_in[12];
    const float* pw1 = (const float*)d_in[13]; const float* pb1 = (const float*)d_in[14];
    const float* pw2 = (const float*)d_in[15]; const float* pb2 = (const float*)d_in[16];
    float* out = (float*)d_out;

    char* ws = (char*)d_ws;
    float*          h0 = (float*)(ws + 0);              // 512*512 f32  (1 MB)
    float*          h1 = (float*)(ws + 1048576);        // 512*256 f32
    float*          h2 = (float*)(ws + 1572864);        // 512*64  f32
    float*          z  = (float*)(ws + 1703936);        // 512*432 f32
    unsigned short* zb = (unsigned short*)(ws + 2588672); // 512*448 bf16 (16B aligned)
    float*          C  = (float*)(ws + 3047424);        // 512*512 f32 (p0 in place)
    float*          p1 = (float*)(ws + 4096000);        // 512*256 f32

    // dense MLP (tiny)
    k_linrelu<<<(BATCH*512 + 255)/256, 256, 0, stream>>>(dense, dw0, db0, h0, BATCH, 13, 512, 1);
    k_linrelu<<<(BATCH*256 + 255)/256, 256, 0, stream>>>(h0, dw1, db1, h1, BATCH, 512, 256, 1);
    k_linrelu<<<(BATCH*64  + 255)/256, 256, 0, stream>>>(h1, dw2, db2, h2, BATCH, 256, 64, 1);

    // concat: dense tail layer + embedding gathers, fp32 + bf16(padded)
    k_concat<<<(BATCH*KPAD + 255)/256, 256, 0, stream>>>(h2, dw3, db3, sparse, emb, z, zb);

    // big interaction GEMM (97.8 GFLOP) via bf16 WMMA, i-split atomically reduced
    k_zero<<<(BATCH*NPRED0 + 255)/256, 256, 0, stream>>>(C, BATCH*NPRED0);
    dim3 grid(NPRED0/64, BATCH/128, KCAT/16);   // (8, 4, 27)
    k_interact<<<grid, 256, 0, stream>>>(zb, z, pw0, C);

    // prediction MLP tail
    k_biasrelu<<<(BATCH*NPRED0 + 255)/256, 256, 0, stream>>>(C, pb0, BATCH, NPRED0);
    k_linrelu<<<(BATCH*256 + 255)/256, 256, 0, stream>>>(C, pw1, pb1, p1, BATCH, 512, 256, 1);
    k_final<<<(BATCH + 255)/256, 256, 0, stream>>>(p1, pw2, pb2, out);
}
